// MotionTransformer_13726715478597
// MI455X (gfx1250) — matrix operand order repaired and checked by reading, not compiled, not run on hardware
//
#include <hip/hip_runtime.h>

// ---------------- model dims ----------------
#define B       32
#define T       196
#define BT      (B*T)          // 6272
#define D       512
#define H       8
#define DH      64
#define TE      2048
#define FFND    1024
#define TXT     256
#define NTXT    77
#define BN      (B*NTXT)       // 2464
#define INF     263
#define INFP    288            // 263 padded to multiple of 32

typedef _Float16 v16h __attribute__((ext_vector_type(16)));
typedef float    v8f  __attribute__((ext_vector_type(8)));

#define ACT_NONE 0
#define ACT_GELU 1

__device__ __forceinline__ float geluf(float x) {
    float x3 = x * x * x;
    return 0.5f * x * (1.0f + tanhf(0.7978845608028654f * (x + 0.044715f * x3)));
}
__device__ __forceinline__ float siluf(float x) { return x / (1.0f + __expf(-x)); }

// ---------------- WMMA GEMM ----------------
// C[M,N] = act(A[M,Kp] @ Wt[Np,Kp]^T + bias) + resid
// Wt is the transposed+converted weight: Wt[n,k] = (half)W[k,n], zero padded.
// 4 waves per block; each wave computes a 16x128 tile of C with 8 accumulators,
// reusing one A fragment (and its f32->f16 converts) per K-step
// -> 8 x v_wmma_f32_16x16x32_f16 per iteration.
__global__ __launch_bounds__(128) void mt_gemm(
    const float* __restrict__ A, int Kp,
    const _Float16* __restrict__ Wt,
    const float* __restrict__ bias,
    const float* __restrict__ resid,
    float* __restrict__ C, int M, int N, int act)
{
    const int lane = threadIdx.x & 31;
    const int wave = threadIdx.x >> 5;
    const int mt = blockIdx.y * 4 + wave;
    if (mt * 16 >= M) return;              // wave-uniform: EXEC stays all-ones
    const int l16 = lane & 15;
    const int hlf = lane >> 4;             // 0 or 1
    const int m  = mt * 16 + l16;
    const int n0 = blockIdx.x * 128 + l16;

    const float* __restrict__ arow = A + (size_t)m * Kp;

    v8f acc[8];
#pragma unroll
    for (int t = 0; t < 8; ++t)
        acc[t] = (v8f){0.f, 0.f, 0.f, 0.f, 0.f, 0.f, 0.f, 0.f};

    for (int kk = 0; kk < Kp; kk += 32) {
        // Pull next K-step lines toward the WGP while WMMAs execute
        // (speculative prefetch: OOB at the last iter is silently dropped).
        __builtin_prefetch(arow + kk + 32, 0, 1);
        __builtin_prefetch(Wt + (size_t)n0 * Kp + kk + 32, 0, 1);

        // A fragment (16-bit A 16x32, contiguous K to match B): lanes 0-15 K 0..15,
        // lanes 16-31 K 16..31.
        const int ka = kk + hlf * 16;
        float4 f0 = *(const float4*)(arow + ka);
        float4 f1 = *(const float4*)(arow + ka + 4);
        float4 f2 = *(const float4*)(arow + ka + 8);
        float4 f3 = *(const float4*)(arow + ka + 12);
        v16h a;
        a[0]  = (_Float16)f0.x; a[1]  = (_Float16)f0.y;
        a[2]  = (_Float16)f0.z; a[3]  = (_Float16)f0.w;
        a[4]  = (_Float16)f1.x; a[5]  = (_Float16)f1.y;
        a[6]  = (_Float16)f1.z; a[7]  = (_Float16)f1.w;
        a[8]  = (_Float16)f2.x; a[9]  = (_Float16)f2.y;
        a[10] = (_Float16)f2.z; a[11] = (_Float16)f2.w;
        a[12] = (_Float16)f3.x; a[13] = (_Float16)f3.y;
        a[14] = (_Float16)f3.z; a[15] = (_Float16)f3.w;

        // B fragment (16-bit B 32x16, contiguous K to match A): lanes 0-15 K 0..15, lanes 16-31 K 16..31,
        // one column (N) per lane -> contiguous 32B load from transposed weights.
        const int kb = kk + hlf * 16;
#pragma unroll
        for (int t = 0; t < 8; ++t) {
            v16h bfrag = *(const v16h*)(Wt + (size_t)(n0 + t * 16) * Kp + kb);
            acc[t] = __builtin_amdgcn_wmma_f32_16x16x32_f16(
                false, a, false, bfrag, (short)0, acc[t], false, false);
        }
    }

    // C layout: col = lane%16, row = vgpr + 8*(lane>=16)
#pragma unroll
    for (int t = 0; t < 8; ++t) {
        const int n = n0 + t * 16;
        const float bs = (bias != nullptr && n < N) ? bias[n] : 0.f;
#pragma unroll
        for (int r = 0; r < 8; ++r) {
            const int mr = mt * 16 + hlf * 8 + r;
            if (n < N && mr < M) {
                float c = acc[t][r] + bs;
                if (act == ACT_GELU) c = geluf(c);
                if (resid != nullptr) c += resid[(size_t)mr * N + n];
                C[(size_t)mr * N + n] = c;
            }
        }
    }
}

// W[K,N] f32 -> Wt[Np,Kp] f16 (zero padded)
__global__ void mt_transpose_f2h(const float* __restrict__ W, _Float16* __restrict__ Wt,
                                 int K, int N, int Kp, int Np)
{
    int k = blockIdx.x * blockDim.x + threadIdx.x;
    int n = blockIdx.y;
    if (k >= Kp || n >= Np) return;
    float v = (k < K && n < N) ? W[(size_t)k * N + n] : 0.f;
    Wt[(size_t)n * Kp + k] = (_Float16)v;
}

// ---------------- LayerNorm (wave per row) ----------------
__global__ __launch_bounds__(128) void mt_layernorm(
    const float* __restrict__ x, const float* __restrict__ g,
    const float* __restrict__ b, float* __restrict__ out, int Dn, int rows)
{
    const int wave = threadIdx.x >> 5, lane = threadIdx.x & 31;
    const int row = blockIdx.x * 4 + wave;
    if (row >= rows) return;
    const float* xr = x + (size_t)row * Dn;
    float s = 0.f, s2 = 0.f;
    for (int i = lane; i < Dn; i += 32) { float v = xr[i]; s += v; s2 += v * v; }
    for (int m = 16; m; m >>= 1) { s += __shfl_xor(s, m, 32); s2 += __shfl_xor(s2, m, 32); }
    const float mean = s / (float)Dn;
    const float inv  = rsqrtf(s2 / (float)Dn - mean * mean + 1e-5f);
    float* o = out + (size_t)row * Dn;
    for (int i = lane; i < Dn; i += 32) o[i] = (xr[i] - mean) * inv * g[i] + b[i];
}

// LN then *(1+scale)+shift with per-batch scale/shift from eo[b,0:512]/[512:1024]
__global__ __launch_bounds__(128) void mt_ln_stylize(
    const float* __restrict__ x, const float* __restrict__ g,
    const float* __restrict__ b, const float* __restrict__ eo,
    float* __restrict__ out, int Tn, int rows)
{
    const int wave = threadIdx.x >> 5, lane = threadIdx.x & 31;
    const int row = blockIdx.x * 4 + wave;
    if (row >= rows) return;
    const float* xr = x + (size_t)row * D;
    const float* e  = eo + (size_t)(row / Tn) * (2 * D);
    float s = 0.f, s2 = 0.f;
    for (int i = lane; i < D; i += 32) { float v = xr[i]; s += v; s2 += v * v; }
    for (int m = 16; m; m >>= 1) { s += __shfl_xor(s, m, 32); s2 += __shfl_xor(s2, m, 32); }
    const float mean = s * (1.f / D);
    const float inv  = rsqrtf(s2 * (1.f / D) - mean * mean + 1e-5f);
    float* o = out + (size_t)row * D;
    for (int i = lane; i < D; i += 32)
        o[i] = ((xr[i] - mean) * inv * g[i] + b[i]) * (1.f + e[i]) + e[D + i];
}

// ---------------- elementwise ----------------
__global__ void mt_silu(const float* __restrict__ x, float* __restrict__ o, int n)
{
    int i = blockIdx.x * blockDim.x + threadIdx.x;
    if (i < n) o[i] = siluf(x[i]);
}

__global__ void mt_timestep(const int* __restrict__ ts, float* __restrict__ temb)
{
    int idx = blockIdx.x * blockDim.x + threadIdx.x;   // B*512
    if (idx >= B * D) return;
    int b0 = idx >> 9, i = idx & 511;
    float t = (float)ts[b0];
    int j = (i < 256) ? i : i - 256;
    float freq = __expf(-logf(10000.f) * (float)j / 256.f);
    float arg = t * freq;
    temb[idx] = (i < 256) ? cosf(arg) : sinf(arg);
}

__global__ void mt_pad_x(const float* __restrict__ x, float* __restrict__ xp, int n)
{
    int idx = blockIdx.x * blockDim.x + threadIdx.x;   // BT*INFP
    if (idx >= n) return;
    int row = idx / INFP, c = idx % INFP;
    xp[idx] = (c < INF) ? x[(size_t)row * INF + c] : 0.f;
}

__global__ void mt_add_seq(float* __restrict__ h, const float* __restrict__ seq, int n)
{
    int idx = blockIdx.x * blockDim.x + threadIdx.x;   // BT*512
    if (idx >= n) return;
    int t = (idx >> 9) % T, c = idx & 511;
    h[idx] += seq[(size_t)t * D + c];
}

// mode 0: k += (1-mask)*-1e6 ; mode 1: v *= mask    (mask: [B*T])
__global__ void mt_apply_mask(float* __restrict__ p, const float* __restrict__ mask,
                              int n, int mode)
{
    int idx = blockIdx.x * blockDim.x + threadIdx.x;   // BT*512
    if (idx >= n) return;
    float mk = mask[idx >> 9];
    if (mode == 0) p[idx] += (1.f - mk) * -1e6f;
    else           p[idx] *= mk;
}

// softmax over the per-head 64 chunk of a 512-wide row (axis=-1)
__global__ void mt_softmax_last64(float* __restrict__ x, int rows)
{
    int idx = blockIdx.x * blockDim.x + threadIdx.x;   // rows*8
    if (idx >= rows * H) return;
    float* p = x + (size_t)(idx >> 3) * D + (size_t)(idx & 7) * DH;
    float mx = -1e30f;
    for (int i = 0; i < DH; ++i) mx = fmaxf(mx, p[i]);
    float s = 0.f;
    for (int i = 0; i < DH; ++i) { float e = __expf(p[i] - mx); p[i] = e; s += e; }
    float r = 1.f / s;
    for (int i = 0; i < DH; ++i) p[i] *= r;
}

// softmax over axis=1 (tokens) of x[Bn,Tn,C] for each (b,c)
__global__ void mt_softmax_axis1(float* __restrict__ x, int Bn, int Tn, int C)
{
    int idx = blockIdx.x * blockDim.x + threadIdx.x;   // Bn*C
    if (idx >= Bn * C) return;
    int b0 = idx / C, c = idx % C;
    float* p = x + (size_t)b0 * Tn * C + c;
    float mx = -1e30f;
    for (int t = 0; t < Tn; ++t) mx = fmaxf(mx, p[(size_t)t * C]);
    float s = 0.f;
    for (int t = 0; t < Tn; ++t) { float e = __expf(p[(size_t)t * C] - mx); p[(size_t)t * C] = e; s += e; }
    float r = 1.f / s;
    for (int t = 0; t < Tn; ++t) p[(size_t)t * C] *= r;
}

// att[b,h,d,l] = sum_t k[b,t,h*64+d] * v[b,t,h*64+l]
__global__ void mt_attn_kv(const float* __restrict__ k, const float* __restrict__ v,
                           float* __restrict__ att, int Tn)
{
    int idx = blockIdx.x * blockDim.x + threadIdx.x;   // B*8*64*64
    if (idx >= B * H * DH * DH) return;
    int l = idx & 63, d = (idx >> 6) & 63, h0 = (idx >> 12) & 7, b0 = idx >> 15;
    const float* kp = k + (size_t)b0 * Tn * D + h0 * DH + d;
    const float* vp = v + (size_t)b0 * Tn * D + h0 * DH + l;
    float s = 0.f;
    for (int t = 0; t < Tn; ++t) s += kp[(size_t)t * D] * vp[(size_t)t * D];
    att[idx] = s;
}

// y[row,h*64+l] = sum_d q[row,h*64+d] * att[b,h,d,l]   (row = b*T+t)
__global__ void mt_attn_q(const float* __restrict__ q, const float* __restrict__ att,
                          float* __restrict__ y, int rows, int Tn)
{
    int idx = blockIdx.x * blockDim.x + threadIdx.x;   // rows*512
    if (idx >= rows * D) return;
    int l = idx & 63, h0 = (idx >> 6) & 7, row = idx >> 9, b0 = row / Tn;
    const float* qp = q + (size_t)row * D + h0 * DH;
    const float* ap = att + (((size_t)b0 * H + h0) * DH) * DH + l;
    float s = 0.f;
    for (int d = 0; d < DH; ++d) s += qp[d] * ap[(size_t)d * DH];
    y[idx] = s;
}

// ---------------- orchestration ----------------
extern "C" void kernel_launch(void* const* d_in, const int* in_sizes, int n_in,
                              void* d_out, int out_size, void* d_ws, size_t ws_size,
                              hipStream_t stream)
{
    (void)in_sizes; (void)n_in; (void)out_size; (void)ws_size;
    const float* X    = (const float*)d_in[0];
    const int*   TS   = (const int*)d_in[1];
    const float* MASK = (const float*)d_in[2];
    const float* XF   = (const float*)d_in[3];
    auto P = [&](int i) { return (const float*)d_in[i]; };
    const float* in_w = P(4);  const float* in_b = P(5);  const float* seq = P(6);
    const float* tw1  = P(7);  const float* tb1  = P(8);
    const float* tw2  = P(9);  const float* tb2  = P(10);
    const float* outw = P(331); const float* outb = P(332);

    // workspace bump allocator (256B aligned)
    char* base = (char*)d_ws; size_t off = 0;
    auto alloc = [&](size_t bytes) { void* p = base + off; off = (off + bytes + 255) & ~(size_t)255; return p; };
    float* temb = (float*)alloc((size_t)B * D * 4);
    float* emb1 = (float*)alloc((size_t)B * TE * 4);
    float* emb  = (float*)alloc((size_t)B * TE * 4);
    float* sile = (float*)alloc((size_t)B * TE * 4);
    float* eo   = (float*)alloc((size_t)B * 2 * D * 4);
    float* xpad = (float*)alloc((size_t)BT * INFP * 4);
    float* hbuf = (float*)alloc((size_t)BT * D * 4);
    float* xn   = (float*)alloc((size_t)BT * D * 4);
    float* bufQ = (float*)alloc((size_t)BT * D * 4);
    float* bufK = (float*)alloc((size_t)BT * D * 4);
    float* bufV = (float*)alloc((size_t)BT * D * 4);
    float* bufY = (float*)alloc((size_t)BT * D * 4);
    float* mid  = (float*)alloc((size_t)BT * FFND * 4);
    float* att  = (float*)alloc((size_t)B * H * DH * DH * 4);
    float* xfn  = (float*)alloc((size_t)BN * TXT * 4);
    float* ck   = (float*)alloc((size_t)BN * D * 4);
    float* cv   = (float*)alloc((size_t)BN * D * 4);
    _Float16* wt = (_Float16*)alloc((size_t)2048 * 2048 * 2);
    float* h2 = bufQ;   // reuse (Q/K dead by stylization time)
    float* s2 = bufK;

    auto gemmTW = [&](const float* A, const float* W, const float* bias, const float* resid,
                      float* C, int M, int K, int N, int Kp, int Np, int act) {
        dim3 tg((Kp + 255) / 256, Np);
        mt_transpose_f2h<<<tg, 256, 0, stream>>>(W, wt, K, N, Kp, Np);
        dim3 gg(Np / 128, (M + 63) / 64);
        mt_gemm<<<gg, 128, 0, stream>>>(A, Kp, wt, bias, resid, C, M, N, act);
    };
    auto styl = [&](const float* yin, int pb) { // pb: we,be,ln_g,ln_b,wo,bo
        gemmTW(sile, P(pb), P(pb + 1), nullptr, eo, B, TE, 2 * D, TE, 2 * D, ACT_NONE);
        mt_ln_stylize<<<(BT + 3) / 4, 128, 0, stream>>>(yin, P(pb + 2), P(pb + 3), eo, h2, T, BT);
        mt_silu<<<(BT * D + 255) / 256, 256, 0, stream>>>(h2, s2, BT * D);
        gemmTW(s2, P(pb + 4), P(pb + 5), hbuf, hbuf, BT, D, D, D, D, ACT_NONE);
    };

    // ---- time embedding MLP ----
    mt_timestep<<<(B * D + 255) / 256, 256, 0, stream>>>(TS, temb);
    gemmTW(temb, tw1, tb1, nullptr, emb1, B, D, TE, D, TE, ACT_NONE);
    mt_silu<<<(B * TE + 255) / 256, 256, 0, stream>>>(emb1, emb1, B * TE);
    gemmTW(emb1, tw2, tb2, nullptr, emb, B, TE, TE, TE, TE, ACT_NONE);
    mt_silu<<<(B * TE + 255) / 256, 256, 0, stream>>>(emb, sile, B * TE);

    // ---- input projection + positional ----
    mt_pad_x<<<(BT * INFP + 255) / 256, 256, 0, stream>>>(X, xpad, BT * INFP);
    gemmTW(xpad, in_w, in_b, nullptr, hbuf, BT, INF, D, INFP, D, ACT_NONE);
    mt_add_seq<<<(BT * D + 255) / 256, 256, 0, stream>>>(hbuf, seq, BT * D);

    // ---- layers ----
    for (int l = 0; l < 8; ++l) {
        const int lb = 11 + l * 40;
        // --- self attention ---
        mt_layernorm<<<(BT + 3) / 4, 128, 0, stream>>>(hbuf, P(lb + 0), P(lb + 1), xn, D, BT);
        gemmTW(xn, P(lb + 2), P(lb + 3), nullptr, bufQ, BT, D, D, D, D, ACT_NONE);
        gemmTW(xn, P(lb + 4), P(lb + 5), nullptr, bufK, BT, D, D, D, D, ACT_NONE);
        gemmTW(xn, P(lb + 6), P(lb + 7), nullptr, bufV, BT, D, D, D, D, ACT_NONE);
        mt_apply_mask<<<(BT * D + 255) / 256, 256, 0, stream>>>(bufK, MASK, BT * D, 0);
        mt_apply_mask<<<(BT * D + 255) / 256, 256, 0, stream>>>(bufV, MASK, BT * D, 1);
        mt_softmax_last64<<<(BT * H + 255) / 256, 256, 0, stream>>>(bufQ, BT);
        mt_softmax_axis1<<<(B * D + 255) / 256, 256, 0, stream>>>(bufK, B, T, D);
        mt_attn_kv<<<(B * H * DH * DH + 255) / 256, 256, 0, stream>>>(bufK, bufV, att, T);
        mt_attn_q<<<(BT * D + 255) / 256, 256, 0, stream>>>(bufQ, att, bufY, BT, T);
        styl(bufY, lb + 8);
        // --- cross attention ---
        const int cb = lb + 14;
        mt_layernorm<<<(BT + 3) / 4, 128, 0, stream>>>(hbuf, P(cb + 0), P(cb + 1), xn, D, BT);
        mt_layernorm<<<(BN + 3) / 4, 128, 0, stream>>>(XF, P(cb + 2), P(cb + 3), xfn, TXT, BN);
        gemmTW(xn,  P(cb + 4), P(cb + 5), nullptr, bufQ, BT, D,   D, D,   D, ACT_NONE);
        gemmTW(xfn, P(cb + 6), P(cb + 7), nullptr, ck,   BN, TXT, D, TXT, D, ACT_NONE);
        gemmTW(xfn, P(cb + 8), P(cb + 9), nullptr, cv,   BN, TXT, D, TXT, D, ACT_NONE);
        mt_softmax_last64<<<(BT * H + 255) / 256, 256, 0, stream>>>(bufQ, BT);
        mt_softmax_axis1<<<(B * D + 255) / 256, 256, 0, stream>>>(ck, B, NTXT, D);
        mt_attn_kv<<<(B * H * DH * DH + 255) / 256, 256, 0, stream>>>(ck, cv, att, NTXT);
        mt_attn_q<<<(BT * D + 255) / 256, 256, 0, stream>>>(bufQ, att, bufY, BT, T);
        styl(bufY, cb + 10);
        // --- FFN ---
        const int fb = lb + 30;
        gemmTW(hbuf, P(fb + 0), P(fb + 1), nullptr, mid,  BT, D,    FFND, D,    FFND, ACT_GELU);
        gemmTW(mid,  P(fb + 2), P(fb + 3), nullptr, bufY, BT, FFND, D,    FFND, D,    ACT_NONE);
        styl(bufY, fb + 4);
    }

    // ---- output projection (N=263, Np=384 padded, guarded stores) ----
    gemmTW(hbuf, outw, outb, nullptr, (float*)d_out, BT, D, INF, D, 384, ACT_NONE);
}
